// VMATDosePredictor_72739566125243
// MI455X (gfx1250) — matrix operand order repaired and checked by reading, not compiled, not run on hardware
//
#include <hip/hip_runtime.h>

// ---------------------------------------------------------------------------
// Types for CDNA5 WMMA (wave32): A/B = v16h (16 halves / lane), C/D = v8f.
// ---------------------------------------------------------------------------
typedef __attribute__((ext_vector_type(16))) _Float16 half16;
typedef __attribute__((ext_vector_type(8)))  _Float16 half8;
typedef __attribute__((ext_vector_type(8)))  float    floatx8;
typedef __attribute__((ext_vector_type(4)))  int      intx4;
typedef __attribute__((ext_vector_type(8)))  int      intx8;

union H16I8 { half16 h; intx8 i; };

__device__ __forceinline__ half8 h8zero() { half8 z = {}; return z; }

// ---------------------------------------------------------------------------
// Implicit-GEMM convolution (channels-last f16 activations, f32 accumulate).
// One wave computes a 16(Co) x 32(pixel) tile: one A (weights) fragment feeds
// two V_WMMA_F32_16X16X32_F16 ops (two 16-pixel B fragments).
// Weights: f16 [Co][T][Ci] (T = KD*KH*KW taps). Requires Ci%8==0, Co%16==0.
// Shape constants are template parameters so k -> (tap, ci) decomposition is
// shifts/constant-muls; predication is branchless at dword granularity
// (clamp addresses + v_cndmask_b32), keeping EXEC all-ones around the WMMAs.
//
// VGPR layouts (C per CDNA5 ISA 7.12.2; A/B use this code's k order):
//   A (16x32): row M = lane&15;  K = 16 consecutive from k0+16*(lane>=16)
//   B (32x16): col N = lane&15;  K = 16 consecutive from k0+16*(lane>=16)
//   C (16x16): c[r] -> M = r + 8*(lane>=16), N = lane&15
// ---------------------------------------------------------------------------
template <int KD, int KH, int KW, int CI>
__global__ void __launch_bounds__(256)
conv_igemm(const _Float16* __restrict__ x, const _Float16* __restrict__ w,
           const float* __restrict__ bias, _Float16* __restrict__ y,
           int N, int Di, int Hi, int Wi,
           int Do, int Ho, int Wo, int Co,
           int stride, int pd, int ph, int pw, long totalTiles)
{
    constexpr int T    = KD * KH * KW;
    constexpr int KHW  = KH * KW;
    constexpr int Ktot = T * CI;

    const int lane = threadIdx.x & 31;
    long wave = (long)blockIdx.x * (blockDim.x >> 5) + (threadIdx.x >> 5);
    if (wave >= totalTiles) return;                 // whole wave exits together

    const int coTiles = Co >> 4;
    const int wTiles  = (Wo + 31) >> 5;
    long t = wave;
    const int cot = (int)(t % coTiles); t /= coTiles;
    const int wt  = (int)(t % wTiles);  t /= wTiles;
    const int ho  = (int)(t % Ho);      t /= Ho;
    const int dz  = (int)(t % Do);
    const int n   = (int)(t / Do);

    const int p    = lane & 15;
    const int hi16 = lane >> 4;
    const int wo0  = (wt << 5) + p;                 // pixel tile 0
    const int wo1  = wo0 + 16;                      // pixel tile 1
    const int co_a = (cot << 4) + p;                // A-fragment row (Co index)
    const size_t wrow = (size_t)co_a * T;

    const int ibase_h = ho * stride - ph;
    const int ibase_d = dz * stride - pd;

    floatx8 acc0 = {}, acc1 = {};
    for (int k0 = 0; k0 < Ktot; k0 += 32) {
        // -------- A fragment (weights) --------
        H16I8 Au;
        {
#pragma unroll
            for (int gidx = 0; gidx < 2; ++gidx) {
                int kk = k0 + (hi16 << 4) + gidx * 8;
                int tt = kk / CI, ci = kk - tt * CI;       // compile-time div
                int tts = (Ktot % 32) ? (tt >= T ? T - 1 : tt) : tt;
                intx4 g = *(const intx4*)(w + (wrow + tts) * CI + ci);
                bool val = (Ktot % 32 == 0) || (kk < Ktot);
#pragma unroll
                for (int j = 0; j < 4; ++j)
                    Au.i[gidx * 4 + j] = val ? g[j] : 0;
            }
        }
        if (k0 + 32 < Ktot)                           // hint: stream next weights
            __builtin_prefetch(w + wrow * CI + k0 + 32, 0, 1);

        // -------- B fragments (input patches, branchless zero-pad) --------
        H16I8 B0u, B1u;
        {
            int kbase = k0 + (hi16 << 4);
#pragma unroll
            for (int gidx = 0; gidx < 2; ++gidx) {
                int kk = kbase + gidx * 8;
                int tt = kk / CI, ci = kk - tt * CI;       // compile-time div
                int kd = tt / KHW; int r2 = tt - kd * KHW;
                int kh = r2 / KW;  int kw = r2 - kh * KW;
                int id = ibase_d + kd;
                int ih = ibase_h + kh;
                bool vdh = ((Ktot % 32 == 0) || (kk < Ktot)) &&
                           ((unsigned)id < (unsigned)Di) &&
                           ((unsigned)ih < (unsigned)Hi);
                int idc = id < 0 ? 0 : (id >= Di ? Di - 1 : id);
                int ihc = ih < 0 ? 0 : (ih >= Hi ? Hi - 1 : ih);
                const _Float16* row =
                    x + ((((size_t)n * Di + idc) * Hi + ihc) * (size_t)Wi) * CI + ci;

                int iw0 = wo0 * stride - pw + kw;
                int iw1 = wo1 * stride - pw + kw;
                bool v0 = vdh && ((unsigned)iw0 < (unsigned)Wi);
                bool v1 = vdh && ((unsigned)iw1 < (unsigned)Wi);
                int iw0c = iw0 < 0 ? 0 : (iw0 >= Wi ? Wi - 1 : iw0);
                int iw1c = iw1 < 0 ? 0 : (iw1 >= Wi ? Wi - 1 : iw1);
                intx4 g0 = *(const intx4*)(row + (size_t)iw0c * CI);
                intx4 g1 = *(const intx4*)(row + (size_t)iw1c * CI);
#pragma unroll
                for (int j = 0; j < 4; ++j) {
                    B0u.i[gidx * 4 + j] = v0 ? g0[j] : 0;
                    B1u.i[gidx * 4 + j] = v1 ? g1[j] : 0;
                }
            }
        }
        acc0 = __builtin_amdgcn_wmma_f32_16x16x32_f16(false, Au.h, false, B0u.h,
                                                      (short)0, acc0, false, false);
        acc1 = __builtin_amdgcn_wmma_f32_16x16x32_f16(false, Au.h, false, B1u.h,
                                                      (short)0, acc1, false, false);
    }

    const int co0 = (cot << 4) + (hi16 << 3);
    const size_t rowbase = (((size_t)n * Do + dz) * Ho + ho) * (size_t)Wo;
    float bb[8];
#pragma unroll
    for (int r = 0; r < 8; ++r) bb[r] = bias[co0 + r];
    if (wo0 < Wo) {
        half8 o;
#pragma unroll
        for (int r = 0; r < 8; ++r) o[r] = (_Float16)(acc0[r] + bb[r]);
        *(half8*)(y + (rowbase + wo0) * Co + co0) = o;
    }
    if (wo1 < Wo) {
        half8 o;
#pragma unroll
        for (int r = 0; r < 8; ++r) o[r] = (_Float16)(acc1[r] + bb[r]);
        *(half8*)(y + (rowbase + wo1) * Co + co0) = o;
    }
}

// ---------------------------------------------------------------------------
// Direct small-Ci kernels (WMMA would waste K on Ci=1/2).
// ---------------------------------------------------------------------------
// b1c1: f32 NCHW [N][2][560][560] -> f16 [N][280][280][16], k3 s2 p1.
// w: f16 [9][2][16]
__global__ void bev_conv1(const float* __restrict__ x, const _Float16* __restrict__ w,
                          const float* __restrict__ bias, _Float16* __restrict__ y, int N)
{
    const int HO = 280, WO = 280, HI = 560, WI = 560;
    long idx = (long)blockIdx.x * blockDim.x + threadIdx.x;
    long total = (long)N * HO * WO;
    if (idx >= total) return;
    int wo = (int)(idx % WO); long r = idx / WO;
    int ho = (int)(r % HO);   int n = (int)(r / HO);
    float acc[16];
#pragma unroll
    for (int c = 0; c < 16; ++c) acc[c] = bias[c];
#pragma unroll
    for (int kh = 0; kh < 3; ++kh) {
        int ih = ho * 2 - 1 + kh; if ((unsigned)ih >= (unsigned)HI) continue;
#pragma unroll
        for (int kw = 0; kw < 3; ++kw) {
            int iw = wo * 2 - 1 + kw; if ((unsigned)iw >= (unsigned)WI) continue;
            int t = kh * 3 + kw;
#pragma unroll
            for (int ci = 0; ci < 2; ++ci) {
                float xv = x[(((size_t)n * 2 + ci) * HI + ih) * WI + iw];
                const _Float16* wp = w + (t * 2 + ci) * 16;
#pragma unroll
                for (int c = 0; c < 16; ++c) acc[c] += xv * (float)wp[c];
            }
        }
    }
    _Float16* yo = y + (size_t)idx * 16;
#pragma unroll
    for (int c = 0; c < 16; ++c) yo[c] = (_Float16)acc[c];
}

// enc0_1: ct f16 [192^3] (Ci=1) -> [192^3][16], k3 s1 p1. w: f16 [27][1][16]
__global__ void ct_conv1(const _Float16* __restrict__ x, const _Float16* __restrict__ w,
                         const float* __restrict__ bias, _Float16* __restrict__ y)
{
    const int D = 192;
    long idx = (long)blockIdx.x * blockDim.x + threadIdx.x;
    long total = (long)D * D * D;
    if (idx >= total) return;
    int wx = (int)(idx % D); long r = idx / D;
    int hy = (int)(r % D);   int dz = (int)(r / D);
    float acc[16];
#pragma unroll
    for (int c = 0; c < 16; ++c) acc[c] = bias[c];
#pragma unroll
    for (int kd = 0; kd < 3; ++kd) {
        int id = dz - 1 + kd; if ((unsigned)id >= (unsigned)D) continue;
#pragma unroll
        for (int kh = 0; kh < 3; ++kh) {
            int ih = hy - 1 + kh; if ((unsigned)ih >= (unsigned)D) continue;
#pragma unroll
            for (int kw = 0; kw < 3; ++kw) {
                int iw = wx - 1 + kw; if ((unsigned)iw >= (unsigned)D) continue;
                float xv = (float)x[((size_t)id * D + ih) * D + iw];
                const _Float16* wp = w + ((kd * 3 + kh) * 3 + kw) * 16;
#pragma unroll
                for (int c = 0; c < 16; ++c) acc[c] += xv * (float)wp[c];
            }
        }
    }
    _Float16* yo = y + (size_t)idx * 16;
#pragma unroll
    for (int c = 0; c < 16; ++c) yo[c] = (_Float16)acc[c];
}

// ConvTranspose3d k=2,s=2 (non-overlapping): out voxel maps to one input voxel
// and tap t=(z*2+y)*2+x. w: f16 [8][Ci][CO]. Writes into channel slice of y.
template <int CO>
__global__ void convT_k2(const _Float16* __restrict__ x, const _Float16* __restrict__ w,
                         const float* __restrict__ bias, _Float16* __restrict__ y,
                         int Di, int Hi, int Wi, int Ci, int Ctot, int cofs)
{
    int Do = Di * 2, Ho = Hi * 2, Wo = Wi * 2;
    long idx = (long)blockIdx.x * blockDim.x + threadIdx.x;
    long total = (long)Do * Ho * Wo;
    if (idx >= total) return;
    int wx = (int)(idx % Wo); long r = idx / Wo;
    int hy = (int)(r % Ho);   int dz = (int)(r / Ho);
    int t = ((dz & 1) * 2 + (hy & 1)) * 2 + (wx & 1);
    const _Float16* xp = x + (((size_t)(dz >> 1) * Hi + (hy >> 1)) * Wi + (wx >> 1)) * Ci;
    const _Float16* wp = w + (size_t)t * Ci * CO;
    float acc[CO];
#pragma unroll
    for (int c = 0; c < CO; ++c) acc[c] = bias[c];
    for (int ci = 0; ci < Ci; ++ci) {
        float xv = (float)xp[ci];
        const _Float16* wr = wp + ci * CO;
#pragma unroll
        for (int c = 0; c < CO; ++c) acc[c] += xv * (float)wr[c];
    }
    _Float16* yo = y + (size_t)idx * Ctot + cofs;
#pragma unroll
    for (int c = 0; c < CO; ++c) yo[c] = (_Float16)acc[c];
}

// final 1x1x1 conv (Ci=16 -> 1) + ReLU, f32 output.
__global__ void final_conv(const _Float16* __restrict__ x, const float* __restrict__ w,
                           const float* __restrict__ b, float* __restrict__ out, long P)
{
    long idx = (long)blockIdx.x * blockDim.x + threadIdx.x;
    if (idx >= P) return;
    const _Float16* xp = x + (size_t)idx * 16;
    float a = b[0];
#pragma unroll
    for (int c = 0; c < 16; ++c) a += (float)xp[c] * w[c];
    out[idx] = a > 0.f ? a : 0.f;
}

// ---------------------------------------------------------------------------
// Instance-norm (affine=False) + LeakyReLU(0.2): two-stage fp32 stats.
// C must be a power of two <= 256.
// ---------------------------------------------------------------------------
__global__ void __launch_bounds__(256)
inorm_partial(const _Float16* __restrict__ x, float* __restrict__ sum,
              float* __restrict__ ssq, long S, int C)
{
    int n   = blockIdx.x;
    int tid = threadIdx.x;
    int c   = tid & (C - 1);
    int g   = tid / C;
    int G   = blockDim.x / C;
    long groups = (long)G * gridDim.y;
    long gg = (long)blockIdx.y * G + g;
    const _Float16* xp = x + (size_t)n * S * C;
    float s = 0.f, q = 0.f;
    for (long si = gg; si < S; si += groups) {
        float v = (float)xp[si * C + c];
        s += v; q += v * v;
    }
    __shared__ float ls[256], lq[256];
    ls[tid] = s; lq[tid] = q;
    __syncthreads();
    for (int h = blockDim.x >> 1; h >= C; h >>= 1) {
        if (tid < h) { ls[tid] += ls[tid + h]; lq[tid] += lq[tid + h]; }
        __syncthreads();
    }
    if (tid < C) {
        atomicAdd(&sum[n * C + tid], ls[tid]);
        atomicAdd(&ssq[n * C + tid], lq[tid]);
    }
}

__global__ void norm_lrelu(_Float16* __restrict__ x, const float* __restrict__ sum,
                           const float* __restrict__ ssq, long S, int C, long total)
{
    long idx = (long)blockIdx.x * blockDim.x + threadIdx.x;
    if (idx >= total) return;
    int  c = (int)(idx % C);
    long n = (idx / C) / S;
    float m  = sum[n * C + c] / (float)S;
    float v  = ssq[n * C + c] / (float)S - m * m;
    float sc = rsqrtf(v + 1e-5f);
    float xv = ((float)x[idx] - m) * sc;
    x[idx] = (_Float16)(xv >= 0.f ? xv : 0.2f * xv);
}

// ---------------------------------------------------------------------------
// BEV -> volume projection (bilinear grid_sample, align_corners, zeros pad),
// accumulated over an angle chunk into f32 [96^3][32].
// grid: f32 [180][96][96][96][2]; feat: f16 [na][140][140][32]
// ---------------------------------------------------------------------------
__global__ void project_acc(const float* __restrict__ grid, const _Float16* __restrict__ feat,
                            float* __restrict__ beam, int a0, int na)
{
    const long V3 = 96L * 96 * 96;
    long v = (long)blockIdx.x * blockDim.x + threadIdx.x;
    if (v >= V3) return;
    float acc[32];
#pragma unroll
    for (int c = 0; c < 32; ++c) acc[c] = 0.f;
    for (int a = 0; a < na; ++a) {
        const float* g = grid + ((size_t)(a0 + a) * V3 + v) * 2;
        float xf = (g[0] + 1.f) * 0.5f * 139.f;
        float yf = (g[1] + 1.f) * 0.5f * 139.f;
        float x0f = floorf(xf), y0f = floorf(yf);
        float tx = xf - x0f, ty = yf - y0f;
        int x0 = (int)x0f, y0 = (int)y0f;
#pragma unroll
        for (int dy = 0; dy < 2; ++dy) {
            int yi = y0 + dy; if ((unsigned)yi >= 140u) continue;
            float wy = dy ? ty : 1.f - ty;
#pragma unroll
            for (int dx = 0; dx < 2; ++dx) {
                int xi = x0 + dx; if ((unsigned)xi >= 140u) continue;
                float wt = wy * (dx ? tx : 1.f - tx);
                const _Float16* f = feat + (((size_t)a * 140 + yi) * 140 + xi) * 32;
#pragma unroll
                for (int j = 0; j < 4; ++j) {
                    half8 fv = *(const half8*)(f + j * 8);
#pragma unroll
                    for (int k = 0; k < 8; ++k) acc[j * 8 + k] += wt * (float)fv[k];
                }
            }
        }
    }
    float* b = beam + (size_t)v * 32;
#pragma unroll
    for (int c = 0; c < 32; ++c) b[c] += acc[c];
}

// ---------------------------------------------------------------------------
// Utility kernels
// ---------------------------------------------------------------------------
__global__ void zero_f32k(float* p, long n)
{ long i = (long)blockIdx.x * blockDim.x + threadIdx.x; if (i < n) p[i] = 0.f; }

__global__ void cvt_f32_f16(const float* __restrict__ x, _Float16* __restrict__ y, long n)
{ long i = (long)blockIdx.x * blockDim.x + threadIdx.x; if (i < n) y[i] = (_Float16)x[i]; }

// mode 0: [O][I][T] -> [O][T][I] (igemm)   mode 1: [I][O][T] -> [T][I][O] (convT)
// mode 2: [O][I][T] -> [T][I][O] (direct)
__global__ void cvt_w(const float* __restrict__ in, _Float16* __restrict__ out,
                      int O, int I, int T, int mode)
{
    long idx = (long)blockIdx.x * blockDim.x + threadIdx.x;
    long total = (long)O * I * T;
    if (idx >= total) return;
    int t = (int)(idx % T); long r = idx / T;
    int o, i;
    if (mode == 1) { o = (int)(r % O); i = (int)(r / O); }
    else           { i = (int)(r % I); o = (int)(r / I); }
    long oidx = (mode == 0) ? ((long)o * T + t) * I + i
                            : ((long)t * I + i) * O + o;
    out[oidx] = (_Float16)in[idx];
}

__global__ void copy_slice(const _Float16* __restrict__ x, _Float16* __restrict__ y,
                           long P, int C, int Ctot, int cofs)
{
    long idx = (long)blockIdx.x * blockDim.x + threadIdx.x;
    if (idx >= P * C) return;
    long p = idx / C; int c = (int)(idx % C);
    y[p * Ctot + cofs + c] = x[idx];
}

__global__ void beam_to_slice(const float* __restrict__ beam, _Float16* __restrict__ y,
                              int Ctot, int cofs)
{
    const long V3 = 96L * 96 * 96;
    long idx = (long)blockIdx.x * blockDim.x + threadIdx.x;
    if (idx >= V3 * 32) return;
    long v = idx / 32; int c = (int)(idx % 32);
    y[v * Ctot + cofs + c] = (_Float16)beam[idx];
}

// ---------------------------------------------------------------------------
// Host orchestration
// ---------------------------------------------------------------------------
static inline long cdivl(long a, long b) { return (a + b - 1) / b; }

template <int KD, int KH, int KW, int CI>
static void igT(hipStream_t stream, const _Float16* x, const _Float16* w, const float* b,
                _Float16* y, int N, int Di, int Hi, int Wi,
                int Do, int Ho, int Wo, int Co, int st, int pd, int ph, int pw)
{
    long tiles = (long)N * Do * Ho * ((Wo + 31) / 32) * (Co / 16);
    unsigned blocks = (unsigned)cdivl(tiles, 8);
    conv_igemm<KD, KH, KW, CI><<<blocks, 256, 0, stream>>>(
        x, w, b, y, N, Di, Hi, Wi, Do, Ho, Wo, Co, st, pd, ph, pw, tiles);
}

extern "C" void kernel_launch(void* const* d_in, const int* in_sizes, int n_in,
                              void* d_out, int out_size, void* d_ws, size_t ws_size,
                              hipStream_t stream)
{
    (void)in_sizes; (void)n_in; (void)out_size;
    // Input order: ct, bev_apertures, grid, then params in dict insertion order
    // (each conv: _w then _b).
    const float* ct   = (const float*)d_in[0];
    const float* bev  = (const float*)d_in[1];
    const float* grid = (const float*)d_in[2];
    auto P = [&](int i) { return (const float*)d_in[i]; };

    const long V192 = 192L * 192 * 192;
    const long V96  = 96L * 96 * 96;
    const long V48  = 48L * 48 * 48;
    const int  CH   = 15;     // BEV angle chunk (180/15 = 12 chunks)

    // ---- workspace bump allocator (256B aligned) ----
    char* base = (char*)d_ws; size_t off = 0;
    auto alloc = [&](size_t bytes) -> void* {
        size_t a = (off + 255) & ~(size_t)255; off = a + bytes; return base + a;
    };

    // f16 weights
    struct WC { int idx; int O, I, T; int mode; _Float16* dst; };
    WC wc[] = {
        {3,  16,  2,  9, 2, 0}, // b1c1  (direct)
        {5,  16, 16,  9, 0, 0}, // b1c2
        {7,  32, 16,  9, 0, 0}, // b2c1
        {9,  32, 32,  9, 0, 0}, // b2c2
        {11, 16,  1, 27, 2, 0}, // enc0_1 (direct)
        {13, 16, 16, 27, 0, 0}, // enc0_2
        {15, 32, 16,  8, 0, 0}, // down1
        {17, 32, 32, 27, 0, 0}, // enc1_1
        {19, 32, 32, 27, 0, 0}, // enc1_2
        {21, 64, 32,  8, 0, 0}, // down2
        {23, 64, 64, 27, 0, 0}, // enc2_1
        {25, 64, 64, 27, 0, 0}, // enc2_2
        {27, 128, 64, 8, 0, 0}, // down3
        {29, 128, 128, 27, 0, 0}, // bot_1
        {31, 128, 128, 27, 0, 0}, // bot_2
        {33, 64, 128,  8, 1, 0}, // up2 (convT: [I][O][T])
        {35, 64, 128, 27, 0, 0}, // dec2_1
        {37, 64, 64, 27, 0, 0},  // dec2_2
        {39, 32, 64,  8, 1, 0},  // up1
        {41, 32, 96, 27, 0, 0},  // dec1_1
        {43, 32, 32, 27, 0, 0},  // dec1_2
        {45, 16, 32,  8, 1, 0},  // up0
        {47, 16, 32, 27, 0, 0},  // dec0_1
        {49, 16, 16, 27, 0, 0},  // dec0_2
    };
    const int NW = (int)(sizeof(wc) / sizeof(wc[0]));
    for (int i = 0; i < NW; ++i)
        wc[i].dst = (_Float16*)alloc((size_t)wc[i].O * wc[i].I * wc[i].T * 2);

    // activations (channels-last f16), with time-disjoint aliasing
    _Float16* ct16    = (_Float16*)alloc(V192 * 2);
    _Float16* skip0   = (_Float16*)alloc(V192 * 16 * 2);          // also t_dec0
    _Float16* dec0cat = (_Float16*)alloc(V192 * 32 * 2);          // also A0, d0
    _Float16* e1a     = (_Float16*)alloc(V96 * 32 * 2);
    _Float16* e1b     = (_Float16*)alloc(V96 * 32 * 2);
    _Float16* skip1   = (_Float16*)alloc(V96 * 32 * 2);           // also t_dec1
    _Float16* e2a     = (_Float16*)alloc(V48 * 64 * 2);
    _Float16* e2b     = (_Float16*)alloc(V48 * 64 * 2);
    _Float16* skip2   = (_Float16*)alloc(V48 * 64 * 2);           // also t_dec2
    _Float16* bta     = (_Float16*)alloc((long)24 * 24 * 24 * 128 * 2);
    _Float16* btb     = (_Float16*)alloc((long)24 * 24 * 24 * 128 * 2);
    _Float16* btc     = (_Float16*)alloc((long)24 * 24 * 24 * 128 * 2);
    _Float16* dec2cat = (_Float16*)alloc(V48 * 128 * 2);          // also d2
    _Float16* dec1cat = (_Float16*)alloc(V96 * 96 * 2);           // also d1
    float*    beam    = (float*)alloc(V96 * 32 * 4);
    _Float16* x1      = (_Float16*)alloc((size_t)CH * 280 * 280 * 16 * 2);
    _Float16* x1b     = (_Float16*)alloc((size_t)CH * 280 * 280 * 16 * 2);
    _Float16* x2      = (_Float16*)alloc((size_t)CH * 140 * 140 * 32 * 2);
    _Float16* x2b     = (_Float16*)alloc((size_t)CH * 140 * 140 * 32 * 2);
    float*    nstats  = (float*)alloc(2048 * 4);                  // sum | ssq
    float*    nsum = nstats, *nssq = nstats + 1024;

    if (off > ws_size) return;   // workspace too small: do nothing (safe)

    _Float16* A0 = dec0cat;      // enc0_1 output (encoder-time alias)
    _Float16* d0 = dec0cat;      // dec0_2 output (decoder-time alias)

    auto norm = [&](_Float16* xb, int N, long S, int C) {
        zero_f32k<<<8, 256, 0, stream>>>(nstats, 2048);
        long byl = cdivl(S, 2048); int By = byl > 256 ? 256 : (int)byl;
        inorm_partial<<<dim3((unsigned)N, (unsigned)By), 256, 0, stream>>>(xb, nsum, nssq, S, C);
        long total = (long)N * S * C;
        norm_lrelu<<<(unsigned)cdivl(total, 256), 256, 0, stream>>>(xb, nsum, nssq, S, C, total);
    };

    // ---- 1. weight conversions ----
    for (int i = 0; i < NW; ++i) {
        long tot = (long)wc[i].O * wc[i].I * wc[i].T;
        cvt_w<<<(unsigned)cdivl(tot, 256), 256, 0, stream>>>(P(wc[i].idx), wc[i].dst,
                                                             wc[i].O, wc[i].I, wc[i].T, wc[i].mode);
    }
    _Float16 *w_b1c1 = wc[0].dst,  *w_b1c2 = wc[1].dst,  *w_b2c1 = wc[2].dst,  *w_b2c2 = wc[3].dst;
    _Float16 *w_e01  = wc[4].dst,  *w_e02  = wc[5].dst,  *w_dn1  = wc[6].dst;
    _Float16 *w_e11  = wc[7].dst,  *w_e12  = wc[8].dst,  *w_dn2  = wc[9].dst;
    _Float16 *w_e21  = wc[10].dst, *w_e22  = wc[11].dst, *w_dn3  = wc[12].dst;
    _Float16 *w_bt1  = wc[13].dst, *w_bt2  = wc[14].dst, *w_up2  = wc[15].dst;
    _Float16 *w_d21  = wc[16].dst, *w_d22  = wc[17].dst, *w_up1  = wc[18].dst;
    _Float16 *w_d11  = wc[19].dst, *w_d12  = wc[20].dst, *w_up0  = wc[21].dst;
    _Float16 *w_d01  = wc[22].dst, *w_d02  = wc[23].dst;

    // ---- 2. ct -> f16, beam accumulator -> 0 ----
    cvt_f32_f16<<<(unsigned)cdivl(V192, 256), 256, 0, stream>>>(ct, ct16, V192);
    zero_f32k<<<(unsigned)cdivl(V96 * 32, 256), 256, 0, stream>>>(beam, V96 * 32);

    // ---- 3. BEV branch + projection, chunked over angles ----
    for (int a0 = 0; a0 < 180; a0 += CH) {
        bev_conv1<<<(unsigned)cdivl((long)CH * 280 * 280, 256), 256, 0, stream>>>(
            bev + (size_t)a0 * 2 * 560 * 560, w_b1c1, P(4), x1, CH);
        norm(x1, CH, 280L * 280, 16);
        igT<1, 3, 3, 16>(stream, x1, w_b1c2, P(6), x1b, CH, 1, 280, 280, 1, 280, 280, 16, 1, 0, 1, 1);
        norm(x1b, CH, 280L * 280, 16);
        igT<1, 3, 3, 16>(stream, x1b, w_b2c1, P(8), x2, CH, 1, 280, 280, 1, 140, 140, 32, 2, 0, 1, 1);
        norm(x2, CH, 140L * 140, 32);
        igT<1, 3, 3, 32>(stream, x2, w_b2c2, P(10), x2b, CH, 1, 140, 140, 1, 140, 140, 32, 1, 0, 1, 1);
        norm(x2b, CH, 140L * 140, 32);
        project_acc<<<(unsigned)cdivl(V96, 256), 256, 0, stream>>>(grid, x2b, beam, a0, CH);
    }

    // ---- 4. encoder ----
    ct_conv1<<<(unsigned)cdivl(V192, 256), 256, 0, stream>>>(ct16, w_e01, P(12), A0);
    norm(A0, 1, V192, 16);
    igT<3, 3, 3, 16>(stream, A0, w_e02, P(14), skip0, 1, 192, 192, 192, 192, 192, 192, 16, 1, 1, 1, 1);
    norm(skip0, 1, V192, 16);
    igT<2, 2, 2, 16>(stream, skip0, w_dn1, P(16), e1a, 1, 192, 192, 192, 96, 96, 96, 32, 2, 0, 0, 0);
    igT<3, 3, 3, 32>(stream, e1a, w_e11, P(18), e1b, 1, 96, 96, 96, 96, 96, 96, 32, 1, 1, 1, 1);
    norm(e1b, 1, V96, 32);
    igT<3, 3, 3, 32>(stream, e1b, w_e12, P(20), skip1, 1, 96, 96, 96, 96, 96, 96, 32, 1, 1, 1, 1);
    norm(skip1, 1, V96, 32);
    igT<2, 2, 2, 32>(stream, skip1, w_dn2, P(22), e2a, 1, 96, 96, 96, 48, 48, 48, 64, 2, 0, 0, 0);
    igT<3, 3, 3, 64>(stream, e2a, w_e21, P(24), e2b, 1, 48, 48, 48, 48, 48, 48, 64, 1, 1, 1, 1);
    norm(e2b, 1, V48, 64);
    igT<3, 3, 3, 64>(stream, e2b, w_e22, P(26), skip2, 1, 48, 48, 48, 48, 48, 48, 64, 1, 1, 1, 1);
    norm(skip2, 1, V48, 64);
    igT<2, 2, 2, 64>(stream, skip2, w_dn3, P(28), bta, 1, 48, 48, 48, 24, 24, 24, 128, 2, 0, 0, 0);
    igT<3, 3, 3, 128>(stream, bta, w_bt1, P(30), btb, 1, 24, 24, 24, 24, 24, 24, 128, 1, 1, 1, 1);
    norm(btb, 1, 24L * 24 * 24, 128);
    igT<3, 3, 3, 128>(stream, btb, w_bt2, P(32), btc, 1, 24, 24, 24, 24, 24, 24, 128, 1, 1, 1, 1);
    norm(btc, 1, 24L * 24 * 24, 128);

    // ---- 5. decoder ----
    convT_k2<64><<<(unsigned)cdivl(V48, 256), 256, 0, stream>>>(btc, w_up2, P(34), dec2cat,
                                                                24, 24, 24, 128, 128, 0);
    copy_slice<<<(unsigned)cdivl(V48 * 64, 256), 256, 0, stream>>>(skip2, dec2cat, V48, 64, 128, 64);
    igT<3, 3, 3, 128>(stream, dec2cat, w_d21, P(36), skip2, 1, 48, 48, 48, 48, 48, 48, 64, 1, 1, 1, 1);
    norm(skip2, 1, V48, 64);
    igT<3, 3, 3, 64>(stream, skip2, w_d22, P(38), dec2cat, 1, 48, 48, 48, 48, 48, 48, 64, 1, 1, 1, 1);
    norm(dec2cat, 1, V48, 64);   // d2

    convT_k2<32><<<(unsigned)cdivl(V96, 256), 256, 0, stream>>>(dec2cat, w_up1, P(40), dec1cat,
                                                                48, 48, 48, 64, 96, 0);
    copy_slice<<<(unsigned)cdivl(V96 * 32, 256), 256, 0, stream>>>(skip1, dec1cat, V96, 32, 96, 32);
    beam_to_slice<<<(unsigned)cdivl(V96 * 32, 256), 256, 0, stream>>>(beam, dec1cat, 96, 64);
    igT<3, 3, 3, 96>(stream, dec1cat, w_d11, P(42), skip1, 1, 96, 96, 96, 96, 96, 96, 32, 1, 1, 1, 1);
    norm(skip1, 1, V96, 32);
    igT<3, 3, 3, 32>(stream, skip1, w_d12, P(44), dec1cat, 1, 96, 96, 96, 96, 96, 96, 32, 1, 1, 1, 1);
    norm(dec1cat, 1, V96, 32);   // d1

    convT_k2<16><<<(unsigned)cdivl(V192, 256), 256, 0, stream>>>(dec1cat, w_up0, P(46), dec0cat,
                                                                 96, 96, 96, 32, 32, 0);
    copy_slice<<<(unsigned)cdivl(V192 * 16, 256), 256, 0, stream>>>(skip0, dec0cat, V192, 16, 32, 16);
    igT<3, 3, 3, 32>(stream, dec0cat, w_d01, P(48), skip0, 1, 192, 192, 192, 192, 192, 192, 16, 1, 1, 1, 1);
    norm(skip0, 1, V192, 16);
    igT<3, 3, 3, 16>(stream, skip0, w_d02, P(50), d0, 1, 192, 192, 192, 192, 192, 192, 16, 1, 1, 1, 1);
    norm(d0, 1, V192, 16);

    final_conv<<<(unsigned)cdivl(V192, 256), 256, 0, stream>>>(d0, P(51), P(52),
                                                               (float*)d_out, V192);
}
